// PattentionLayer_12532714569895
// MI455X (gfx1250) — compile-verified, matrix operand order checked by reading, not run on hardware
//
#include <hip/hip_runtime.h>
#include <hip/hip_bf16.h>
#include <math.h>

// Problem sizes (from reference)
#define DIM_IN     1024
#define DIM_OUT    1024
#define NUM_TOKENS 4096
#define NUM_ROWS   8192          // 4 * 2048
#define EPS_F      1e-12f
#define SIM_SCALE  0.03125f      // 1024^-0.5

typedef __bf16 bf16_t;
typedef bf16_t v16bf __attribute__((ext_vector_type(16)));
typedef bf16_t v8bf  __attribute__((ext_vector_type(8)));
typedef float  v8f   __attribute__((ext_vector_type(8)));

typedef __attribute__((address_space(3))) bf16_t lds_bf16_t;   // 32-bit LDS pointer

#define LDSTR 40   // LDS row stride in bf16 elements for a 32-wide K tile (16B aligned, spreads banks)

// CDNA5 async global->LDS copy, 16 bytes per lane (tracked by ASYNCcnt).
__device__ __forceinline__ void async_copy_b128(lds_bf16_t* dst, const bf16_t* src) {
    asm volatile("global_load_async_to_lds_b128 %0, %1, off"
                 :: "v"(dst), "v"(src) : "memory");
}

// ---------------- fp32 -> bf16 convert (one-time, memory-bound) ----------------
__global__ __launch_bounds__(256)
void cvt_f32_bf16_kernel(const float* __restrict__ in, bf16_t* __restrict__ out, int n) {
    int i = blockIdx.x * blockDim.x + threadIdx.x;
    int stride = gridDim.x * blockDim.x;
    for (; i < n; i += stride) out[i] = (bf16_t)in[i];
}

// ---------------- sum-of-squares -> 1/max(sqrt(ss), eps) ----------------
__global__ __launch_bounds__(256)
void invnorm_kernel(float* __restrict__ norms) {
    int i = blockIdx.x * blockDim.x + threadIdx.x;
    if (i < NUM_ROWS) {
        float n = sqrtf(norms[i]);
        norms[i] = 1.0f / fmaxf(n, EPS_F);
    }
}

// ---------------- GEMM1: sim = (X * K^T) * scale ; accumulate row sum-of-squares ----------------
// 256 threads (8 waves). Tile 128x128, K-step 32. Double-buffered LDS fed by
// GLOBAL_LOAD_ASYNC_TO_LDS_B128 so the next tile streams in under the WMMAs.
__global__ __launch_bounds__(256)
void gemm1_kernel(const bf16_t* __restrict__ xb, const bf16_t* __restrict__ kb,
                  bf16_t* __restrict__ simb, float* __restrict__ norms) {
    __shared__ alignas(16) bf16_t As[2][128 * LDSTR];
    __shared__ alignas(16) bf16_t Bs[2][128 * LDSTR];

    const int mBase = blockIdx.y * 128;
    const int nBase = blockIdx.x * 128;
    const int tid  = threadIdx.x;
    const int lane = tid & 31;
    const int wid  = tid >> 5;
    const int wm   = wid & 3;      // M sub-block of wave (32 rows)
    const int wn   = wid >> 2;     // N sub-block of wave (64 cols)
    const int hv   = lane >> 4;    // wave half (K-phase of fragment layout)
    const int l16  = lane & 15;

    // global->LDS mapping: each thread moves 16 bf16 (two 16B chunks) of one row per operand
    const int ldRow  = tid >> 1;   // 0..127
    const int ldHalf = tid & 1;

    const bf16_t* srcA = xb + (size_t)(mBase + ldRow) * DIM_IN + ldHalf * 16;
    const bf16_t* srcB = kb + (size_t)(nBase + ldRow) * DIM_IN + ldHalf * 16;
    lds_bf16_t* dstA = (lds_bf16_t*)(&As[0][0]) + ldRow * LDSTR + ldHalf * 16;
    lds_bf16_t* dstB = (lds_bf16_t*)(&Bs[0][0]) + ldRow * LDSTR + ldHalf * 16;

    v8f acc[2][4] = {};

    // prologue: stream first K-tile into buffer 0
    async_copy_b128(dstA,     srcA);
    async_copy_b128(dstA + 8, srcA + 8);
    async_copy_b128(dstB,     srcB);
    async_copy_b128(dstB + 8, srcB + 8);

    const int STEPS = DIM_IN / 32;
    #pragma unroll 1
    for (int i = 0; i < STEPS; ++i) {
        const int p = i & 1;
        if (i + 1 < STEPS) {
            const int boff = (1 - p) * 128 * LDSTR;
            const int koff = (i + 1) * 32;
            async_copy_b128(dstA + boff,     srcA + koff);
            async_copy_b128(dstA + boff + 8, srcA + koff + 8);
            async_copy_b128(dstB + boff,     srcB + koff);
            async_copy_b128(dstB + boff + 8, srcB + koff + 8);
            // 8 in flight; <=4 left means the 4 older (buffer p) loads completed (in-order)
            asm volatile("s_wait_asynccnt 0x4" ::: "memory");
        } else {
            asm volatile("s_wait_asynccnt 0x0" ::: "memory");
        }
        __syncthreads();   // all waves' buffer-p data now in LDS

        v16bf a[2], b[4];
        #pragma unroll
        for (int mi = 0; mi < 2; ++mi) {
            const bf16_t* ptr = &As[p][0] + (wm * 32 + mi * 16 + l16) * LDSTR;
            v8bf lo = *(const v8bf*)(ptr + 8 * hv);        // K = 8h .. 8h+7
            v8bf hi = *(const v8bf*)(ptr + 16 + 8 * hv);   // K = 16+8h .. 16+8h+7
            a[mi] = __builtin_shufflevector(lo, hi, 0,1,2,3,4,5,6,7,8,9,10,11,12,13,14,15);
        }
        #pragma unroll
        for (int ni = 0; ni < 4; ++ni) {
            const bf16_t* ptr = &Bs[p][0] + (wn * 64 + ni * 16 + l16) * LDSTR;
            v8bf lo = *(const v8bf*)(ptr + 8 * hv);
            v8bf hi = *(const v8bf*)(ptr + 16 + 8 * hv);
            b[ni] = __builtin_shufflevector(lo, hi, 0,1,2,3,4,5,6,7,8,9,10,11,12,13,14,15);
        }
        #pragma unroll
        for (int mi = 0; mi < 2; ++mi)
            #pragma unroll
            for (int ni = 0; ni < 4; ++ni)
                acc[mi][ni] = __builtin_amdgcn_wmma_f32_16x16x32_bf16(
                    false, a[mi], false, b[ni], (short)0, acc[mi][ni], false, false);
        __syncthreads();   // readers done before buffer p is overwritten at i+2
    }

    // Epilogue: scale, store sim (bf16), reduce per-row sum of squares.
    #pragma unroll
    for (int mi = 0; mi < 2; ++mi) {
        #pragma unroll
        for (int r = 0; r < 8; ++r) {
            const int rowg = mBase + wm * 32 + mi * 16 + hv * 8 + r;
            float ss = 0.0f;
            #pragma unroll
            for (int ni = 0; ni < 4; ++ni) {
                float v = acc[mi][ni][r] * SIM_SCALE;
                ss += v * v;
                const int colg = nBase + wn * 64 + ni * 16 + l16;
                simb[(size_t)rowg * NUM_TOKENS + colg] = (bf16_t)v;
            }
            #pragma unroll
            for (int off = 8; off >= 1; off >>= 1)
                ss += __shfl_xor(ss, off, 16);
            if (l16 == 0) atomicAdd(&norms[rowg], ss);
        }
    }
}

// ---------------- GEMM2: out = gelu(sim * invnorm[row]) * V ----------------
// A-tile is transformed (normalize + exact-erf gelu) and B-tile transposed, so
// both go through registers; next step's globals are prefetched into registers
// so their latency hides under WMMA + erf VALU work.
__global__ __launch_bounds__(256)
void gemm2_kernel(const bf16_t* __restrict__ simb, const float* __restrict__ invn,
                  const bf16_t* __restrict__ vb, float* __restrict__ out) {
    __shared__ alignas(16) bf16_t As[128 * LDSTR];
    __shared__ alignas(16) bf16_t Bs[128 * LDSTR];

    const int mBase = blockIdx.y * 128;
    const int nBase = blockIdx.x * 128;   // output-feature (dv) base
    const int tid  = threadIdx.x;
    const int lane = tid & 31;
    const int wid  = tid >> 5;
    const int wm   = wid & 3;
    const int wn   = wid >> 2;
    const int hv   = lane >> 4;
    const int l16  = lane & 15;

    const int ldRow  = tid >> 1;
    const int ldHalf = tid & 1;
    const float inv  = invn[mBase + ldRow];

    const int vTok = tid & 31;        // token within 32-wide K step
    const int vDv0 = (tid >> 5) * 16; // 16 output features per thread

    const bf16_t* srcA = simb + (size_t)(mBase + ldRow) * NUM_TOKENS + ldHalf * 16;
    const bf16_t* srcVb = vb + (size_t)vTok * DIM_OUT + nBase + vDv0;

    v8f acc[2][4] = {};

    // register prefetch of step 0
    union pack8 { v8bf v; bf16_t e[8]; };
    pack8 ra0, ra1, rb0, rb1;
    ra0.v = *(const v8bf*)(srcA);
    ra1.v = *(const v8bf*)(srcA + 8);
    rb0.v = *(const v8bf*)(srcVb);
    rb1.v = *(const v8bf*)(srcVb + 8);

    const int STEPS = NUM_TOKENS / 32;
    #pragma unroll 1
    for (int i = 0; i < STEPS; ++i) {
        // transform current registers -> LDS
        pack8 d0, d1;
        #pragma unroll
        for (int j = 0; j < 8; ++j) {
            float x0 = (float)ra0.e[j] * inv;
            float x1 = (float)ra1.e[j] * inv;
            d0.e[j] = (bf16_t)(0.5f * x0 * (1.0f + erff(x0 * 0.70710678118f)));
            d1.e[j] = (bf16_t)(0.5f * x1 * (1.0f + erff(x1 * 0.70710678118f)));
        }
        *(v8bf*)(As + ldRow * LDSTR + ldHalf * 16)     = d0.v;
        *(v8bf*)(As + ldRow * LDSTR + ldHalf * 16 + 8) = d1.v;
        #pragma unroll
        for (int j = 0; j < 8; ++j) {
            Bs[(vDv0 + j)     * LDSTR + vTok] = rb0.e[j];
            Bs[(vDv0 + 8 + j) * LDSTR + vTok] = rb1.e[j];
        }
        __syncthreads();

        // prefetch step i+1 globals (latency hidden under WMMA + erf below)
        if (i + 1 < STEPS) {
            const size_t ka = (size_t)(i + 1) * 32;
            ra0.v = *(const v8bf*)(srcA + ka);
            ra1.v = *(const v8bf*)(srcA + ka + 8);
            rb0.v = *(const v8bf*)(srcVb + ka * DIM_OUT);
            rb1.v = *(const v8bf*)(srcVb + ka * DIM_OUT + 8);
        }

        v16bf a[2], b[4];
        #pragma unroll
        for (int mi = 0; mi < 2; ++mi) {
            const bf16_t* ptr = As + (wm * 32 + mi * 16 + l16) * LDSTR;
            v8bf lo = *(const v8bf*)(ptr + 8 * hv);
            v8bf hi = *(const v8bf*)(ptr + 16 + 8 * hv);
            a[mi] = __builtin_shufflevector(lo, hi, 0,1,2,3,4,5,6,7,8,9,10,11,12,13,14,15);
        }
        #pragma unroll
        for (int ni = 0; ni < 4; ++ni) {
            const bf16_t* ptr = Bs + (wn * 64 + ni * 16 + l16) * LDSTR;
            v8bf lo = *(const v8bf*)(ptr + 8 * hv);
            v8bf hi = *(const v8bf*)(ptr + 16 + 8 * hv);
            b[ni] = __builtin_shufflevector(lo, hi, 0,1,2,3,4,5,6,7,8,9,10,11,12,13,14,15);
        }
        #pragma unroll
        for (int mi = 0; mi < 2; ++mi)
            #pragma unroll
            for (int ni = 0; ni < 4; ++ni)
                acc[mi][ni] = __builtin_amdgcn_wmma_f32_16x16x32_bf16(
                    false, a[mi], false, b[ni], (short)0, acc[mi][ni], false, false);
        __syncthreads();
    }

    #pragma unroll
    for (int mi = 0; mi < 2; ++mi) {
        #pragma unroll
        for (int r = 0; r < 8; ++r) {
            const int rowg = mBase + wm * 32 + mi * 16 + hv * 8 + r;
            #pragma unroll
            for (int ni = 0; ni < 4; ++ni) {
                const int colg = nBase + wn * 64 + ni * 16 + l16;
                out[(size_t)rowg * DIM_OUT + colg] = acc[mi][ni][r];
            }
        }
    }
}

extern "C" void kernel_launch(void* const* d_in, const int* in_sizes, int n_in,
                              void* d_out, int out_size, void* d_ws, size_t ws_size,
                              hipStream_t stream) {
    (void)in_sizes; (void)n_in; (void)out_size; (void)ws_size;
    const float* x  = (const float*)d_in[0];   // [8192, 1024]
    const float* kp = (const float*)d_in[1];   // [4096, 1024]
    const float* vp = (const float*)d_in[2];   // [4096, 1024]
    float* out = (float*)d_out;                // [8192, 1024] f32

    // Workspace layout (~96.1 MiB):
    char* ws = (char*)d_ws;
    bf16_t* xb   = (bf16_t*)ws;                                     // 16 MiB
    bf16_t* kb   = xb + (size_t)NUM_ROWS * DIM_IN;                  //  8 MiB
    bf16_t* vb   = kb + (size_t)NUM_TOKENS * DIM_IN;                //  8 MiB
    bf16_t* simb = vb + (size_t)NUM_TOKENS * DIM_OUT;               // 64 MiB (fits in 192MB L2)
    float*  norms = (float*)(simb + (size_t)NUM_ROWS * NUM_TOKENS); // 32 KiB

    hipMemsetAsync(norms, 0, NUM_ROWS * sizeof(float), stream);

    cvt_f32_bf16_kernel<<<2048, 256, 0, stream>>>(x,  xb, NUM_ROWS * DIM_IN);
    cvt_f32_bf16_kernel<<<1024, 256, 0, stream>>>(kp, kb, NUM_TOKENS * DIM_IN);
    cvt_f32_bf16_kernel<<<1024, 256, 0, stream>>>(vp, vb, NUM_TOKENS * DIM_OUT);

    dim3 g1(NUM_TOKENS / 128, NUM_ROWS / 128);   // 32 x 64
    gemm1_kernel<<<g1, 256, 0, stream>>>(xb, kb, simb, norms);

    invnorm_kernel<<<NUM_ROWS / 256, 256, 0, stream>>>(norms);

    dim3 g2(DIM_OUT / 128, NUM_ROWS / 128);      // 8 x 64
    gemm2_kernel<<<g2, 256, 0, stream>>>(simb, norms, vb, out);
}
